// NeuronGPTOSSBlock_50998441672786
// MI455X (gfx1250) — compile-verified, matrix operand order checked by reading, not run on hardware
//
#include <hip/hip_runtime.h>

#define NH   32
#define NKV  8
#define HD   64
#define HDIM 2048
#define IDIM 1024
#define NEXP 16
#define SEQ  1024
#define TILES_M (SEQ / 64)                 // 16 row-tiles per expert
#define SLOT_CAP (2 * SEQ + NEXP * 64)     // top-2 assignments, 64-padded

typedef __bf16 bf16;
typedef __attribute__((ext_vector_type(16))) __bf16 v16bf;
typedef __attribute__((ext_vector_type(8)))  __bf16 v8bf;
typedef __attribute__((ext_vector_type(8)))  float  v8f;

#if defined(__has_builtin)
#if __has_builtin(__builtin_amdgcn_tensor_load_to_lds) && \
    __has_builtin(__builtin_amdgcn_s_wait_tensorcnt)
#define HAVE_TDM 1
#endif
#endif
#ifndef HAVE_TDM
#define HAVE_TDM 0
#endif

#if HAVE_TDM
typedef unsigned int u32x4 __attribute__((ext_vector_type(4)));
typedef int          i32x8 __attribute__((ext_vector_type(8)));
typedef int          i32x4 __attribute__((ext_vector_type(4)));
#endif

// ---------------------------------------------------------------- WMMA core
__device__ inline v8f wmma_bf16(v16bf a, v16bf b, v8f c) {
  return __builtin_amdgcn_wmma_f32_16x16x32_bf16(false, a, false, b, (short)0, c,
                                                 false, false);
}

// Load a 16-row fragment (A or B layout) from LDS with given row stride:
// two contiguous 16B ds_load_b128 per lane.
__device__ inline v16bf frag16s(const bf16* base, int row, int half, int stride) {
  const v8bf lo = *(const v8bf*)(base + row * stride + half * 8);
  const v8bf hi = *(const v8bf*)(base + row * stride + 16 + half * 8);
  v16bf r;
#pragma unroll
  for (int i = 0; i < 8; ++i) { r[i] = lo[i]; r[i + 8] = hi[i]; }
  return r;
}

// 8 contiguous fp32 -> v8bf (two global_load_b128 + packed cvt)
__device__ inline v8bf load8_bf(const float* __restrict__ p) {
  const float4 a = *(const float4*)p;
  const float4 b = *(const float4*)(p + 4);
  v8bf r;
  r[0] = (bf16)a.x; r[1] = (bf16)a.y; r[2] = (bf16)a.z; r[3] = (bf16)a.w;
  r[4] = (bf16)b.x; r[5] = (bf16)b.y; r[6] = (bf16)b.z; r[7] = (bf16)b.w;
  return r;
}

// ---------------------------------------------------------------- RMSNorm
__global__ __launch_bounds__(256)
void rmsnorm_kernel(const float* __restrict__ x, const float* __restrict__ w,
                    float* __restrict__ y, int n) {
  __shared__ float red[256];
  const int row = blockIdx.x;
  const float* xr = x + (size_t)row * n;
  float ss = 0.f;
  for (int i = threadIdx.x; i < n; i += 256) { float t = xr[i]; ss += t * t; }
  red[threadIdx.x] = ss;
  __syncthreads();
  for (int s = 128; s > 0; s >>= 1) {
    if (threadIdx.x < s) red[threadIdx.x] += red[threadIdx.x + s];
    __syncthreads();
  }
  const float inv = rsqrtf(red[0] / (float)n + 1e-5f);
  for (int i = threadIdx.x; i < n; i += 256)
    y[(size_t)row * n + i] = xr[i] * inv * w[i];
}

// --------------------------------------------- 64x64 bf16-WMMA GEMM (aligned)
__global__ __launch_bounds__(128)
void gemm64(const float* __restrict__ A, int lda,
            const float* __restrict__ B, int ldb,
            const float* __restrict__ bias, const float* __restrict__ residual,
            float* __restrict__ C, float* __restrict__ C2, int ldc, int K) {
  __shared__ bf16 As[64][32];   // As[m][k]
  __shared__ bf16 Bs[64][32];   // Bs[n][k]
  const int tid = threadIdx.x, wid = tid >> 5, lane = tid & 31;
  const int lr = lane & 15, lh = lane >> 4;
  const int m0 = blockIdx.y * 64, n0 = blockIdx.x * 64;
  const int mw = (wid >> 1) * 32, nw = (wid & 1) * 32;
  v8f acc[2][2] = {};
  for (int k0 = 0; k0 < K; k0 += 32) {
#pragma unroll
    for (int it = 0; it < 2; ++it) {
      const int idx = tid + it * 128;
      const int r = idx >> 2, cq = (idx & 3) * 8;          // A: 64 rows x 32 k
      const float* ap = &A[(size_t)(m0 + r) * lda + k0 + cq];
      *(v8bf*)&As[r][cq] = load8_bf(ap);
      if (k0 + 32 < K) __builtin_prefetch(ap + 32, 0, 0);
      const int kr = idx >> 3, nq = (idx & 7) * 8;         // B: 32 k x 64 n
      const v8bf bb = load8_bf(&B[(size_t)(k0 + kr) * ldb + n0 + nq]);
#pragma unroll
      for (int j = 0; j < 8; ++j) Bs[nq + j][kr] = bb[j];
      if (k0 + 32 < K)
        __builtin_prefetch(&B[(size_t)(k0 + 32 + kr) * ldb + n0 + nq], 0, 0);
    }
    __syncthreads();
    const v16bf a0 = frag16s(&As[mw][0], lr, lh, 32);
    const v16bf a1 = frag16s(&As[mw + 16][0], lr, lh, 32);
    const v16bf b0 = frag16s(&Bs[nw][0], lr, lh, 32);
    const v16bf b1 = frag16s(&Bs[nw + 16][0], lr, lh, 32);
    acc[0][0] = wmma_bf16(a0, b0, acc[0][0]);
    acc[0][1] = wmma_bf16(a0, b1, acc[0][1]);
    acc[1][0] = wmma_bf16(a1, b0, acc[1][0]);
    acc[1][1] = wmma_bf16(a1, b1, acc[1][1]);
    __syncthreads();
  }
#pragma unroll
  for (int mi = 0; mi < 2; ++mi)
#pragma unroll
    for (int ni = 0; ni < 2; ++ni)
#pragma unroll
      for (int v = 0; v < 8; ++v) {
        const int gm = m0 + mw + mi * 16 + v + 8 * lh;
        const int gn = n0 + nw + ni * 16 + lr;
        float o = acc[mi][ni][v];
        if (bias)     o += bias[gn];
        if (residual) o += residual[(size_t)gm * ldc + gn];
        C[(size_t)gm * ldc + gn] = o;
        if (C2) C2[(size_t)gm * ldc + gn] = o;
      }
}

// ---------------------------------------------------------------- RoPE
__global__ void rope_kernel(float* __restrict__ t, int nheads,
                            const int* __restrict__ pos) {
  const int idx = blockIdx.x * blockDim.x + threadIdx.x;
  if (idx >= SEQ * nheads * (HD / 2)) return;
  const int d = idx & 31;
  const int hh = (idx >> 5) % nheads;
  const int s = idx / (32 * nheads);
  const float inv = __expf(-((float)d / 32.f) * 9.210340371976184f); // ln(1e4)
  const float ang = (float)pos[s] * inv;
  const float c = __cosf(ang), si = __sinf(ang);
  const size_t base = (size_t)s * nheads * HD + (size_t)hh * HD;
  const float q0 = t[base + d], q1 = t[base + d + 32];
  t[base + d]      = q0 * c - q1 * si;
  t[base + d + 32] = q1 * c + q0 * si;
}

// -------------------------- Flash attention: 4 q-tiles/block share K/V tiles
__global__ __launch_bounds__(128)
void flash_attn_kernel(const float* __restrict__ Q, const float* __restrict__ Kb,
                       const float* __restrict__ Vb, float* __restrict__ O) {
  __shared__ bf16 Qs[64][64];
  __shared__ bf16 Ks[32][64];
  __shared__ bf16 Vt[64][32];    // transposed: dim x key
  __shared__ bf16 Ps[4][16 * 32];
  const int qblk = blockIdx.x, h = blockIdx.y, hk = h >> 2;  // GQA 4:1
  const int tid = threadIdx.x, wid = tid >> 5, lane = tid & 31;
  const int lr = lane & 15, lh = lane >> 4;
  const int q0r = qblk * 64;
  const int qstride = NH * HD, kstride = NKV * HD;

#pragma unroll
  for (int it = 0; it < 4; ++it) {
    const int idx = tid + it * 128;
    const int r = idx >> 3, cq = (idx & 7) * 8;
    *(v8bf*)&Qs[r][cq] = load8_bf(&Q[(size_t)(q0r + r) * qstride + h * HD + cq]);
  }
  __syncthreads();
  const int qt = qblk * 4 + wid;
  const v16bf aq0 = frag16s(&Qs[wid * 16][0], lr, lh, 64);
  const v16bf aq1 = frag16s(&Qs[wid * 16][32], lr, lh, 64);
  const int mykb = (qt * 16 + 15) >> 5;

  v8f o[4] = {};
  float run_m[8], run_l[8];
#pragma unroll
  for (int v = 0; v < 8; ++v) { run_m[v] = -1e30f; run_l[v] = 0.f; }

  const int kbmax = qblk * 2 + 1;
  for (int kb = 0; kb <= kbmax; ++kb) {
    __syncthreads();
#pragma unroll
    for (int it = 0; it < 2; ++it) {
      const int idx = tid + it * 128;
      const int r = idx >> 3, cq = (idx & 7) * 8;
      *(v8bf*)&Ks[r][cq] =
          load8_bf(&Kb[(size_t)(kb * 32 + r) * kstride + hk * HD + cq]);
      const v8bf vv =
          load8_bf(&Vb[(size_t)(kb * 32 + r) * kstride + hk * HD + cq]);
#pragma unroll
      for (int j = 0; j < 8; ++j) Vt[cq + j][r] = vv[j];
    }
    __syncthreads();
    if (kb > mykb) continue;                 // uniform per wave
    v8f s[2];
#pragma unroll
    for (int j = 0; j < 2; ++j) {
      const v16bf bk0 = frag16s(&Ks[j * 16][0], lr, lh, 64);
      const v16bf bk1 = frag16s(&Ks[j * 16][32], lr, lh, 64);
      v8f t = {};
      t = wmma_bf16(aq0, bk0, t);
      t = wmma_bf16(aq1, bk1, t);
      s[j] = t;
    }
    float p0v[8], p1v[8], alpha[8];
#pragma unroll
    for (int v = 0; v < 8; ++v) {
      const int qr = qt * 16 + v + 8 * lh;
      const int key0 = kb * 32 + lr, key1 = kb * 32 + 16 + lr;
      float s0 = (key0 > qr) ? -1e30f : s[0][v] * 0.125f;
      float s1 = (key1 > qr) ? -1e30f : s[1][v] * 0.125f;
      float mx = fmaxf(s0, s1);
#pragma unroll
      for (int d = 1; d < 16; d <<= 1) mx = fmaxf(mx, __shfl_xor(mx, d));
      const float nm = fmaxf(run_m[v], mx);
      alpha[v] = __expf(run_m[v] - nm);
      const float p0 = __expf(s0 - nm), p1 = __expf(s1 - nm);
      float rs = p0 + p1;
#pragma unroll
      for (int d = 1; d < 16; d <<= 1) rs += __shfl_xor(rs, d);
      run_l[v] = run_l[v] * alpha[v] + rs;
      run_m[v] = nm;
      p0v[v] = p0; p1v[v] = p1;
    }
#pragma unroll
    for (int nt = 0; nt < 4; ++nt)
#pragma unroll
      for (int v = 0; v < 8; ++v) o[nt][v] *= alpha[v];
#pragma unroll
    for (int v = 0; v < 8; ++v) {
      Ps[wid][(v + 8 * lh) * 32 + lr]      = (bf16)p0v[v];
      Ps[wid][(v + 8 * lh) * 32 + 16 + lr] = (bf16)p1v[v];
    }
    const v16bf pA = frag16s(&Ps[wid][0], lr, lh, 32);
#pragma unroll
    for (int nt = 0; nt < 4; ++nt) {
      const v16bf bv = frag16s(&Vt[nt * 16][0], lr, lh, 32);
      o[nt] = wmma_bf16(pA, bv, o[nt]);
    }
  }
#pragma unroll
  for (int nt = 0; nt < 4; ++nt)
#pragma unroll
    for (int v = 0; v < 8; ++v) {
      const int m = v + 8 * lh;
      O[(size_t)(qt * 16 + m) * qstride + h * HD + nt * 16 + lr] =
          o[nt][v] / run_l[v];
    }
}

// ---------------------------------------------------------------- Router
__global__ __launch_bounds__(256)
void router_kernel(const float* __restrict__ X, const float* __restrict__ W,
                   float* __restrict__ logits) {
  const int t = blockIdx.x * 16 + (threadIdx.x >> 4);
  const int e = threadIdx.x & 15;
  float s = 0.f;
  for (int k2 = 0; k2 < HDIM; ++k2)
    s += X[(size_t)t * HDIM + k2] * W[(size_t)k2 * NEXP + e];
  logits[(size_t)t * NEXP + e] = s;
}

// ---------------------------------------------------------------- Routing
__global__ void route_init_kernel(int* counts, int* cursor, int* tokslot,
                                  float* wslot) {
  const int i = blockIdx.x * blockDim.x + threadIdx.x;
  if (i < NEXP) { counts[i] = 0; cursor[i] = 0; }
  if (i < SLOT_CAP) { tokslot[i] = 0; wslot[i] = 0.f; }
}

__global__ void route_topk_kernel(const float* __restrict__ logits,
                                  int* __restrict__ topi, float* __restrict__ topw,
                                  int* __restrict__ counts) {
  const int t = blockIdx.x * blockDim.x + threadIdx.x;
  if (t >= SEQ) return;
  const float* lg = logits + (size_t)t * NEXP;
  float b0 = -1e30f; int i0 = 0;
#pragma unroll
  for (int e = 0; e < NEXP; ++e) { float x = lg[e]; if (x > b0) { b0 = x; i0 = e; } }
  float b1 = -1e30f; int i1 = 0;
#pragma unroll
  for (int e = 0; e < NEXP; ++e) {
    float x = lg[e];
    if (e != i0 && x > b1) { b1 = x; i1 = e; }
  }
  const float e1 = __expf(b1 - b0);
  topi[t * 2] = i0;  topi[t * 2 + 1] = i1;
  topw[t * 2] = 1.f / (1.f + e1);
  topw[t * 2 + 1] = e1 / (1.f + e1);
  atomicAdd(&counts[i0], 1);
  atomicAdd(&counts[i1], 1);
}

__global__ void route_scan_kernel(const int* __restrict__ counts,
                                  int* __restrict__ off, int* __restrict__ pcount) {
  if (threadIdx.x == 0 && blockIdx.x == 0) {
    int run = 0;
    for (int e = 0; e < NEXP; ++e) {
      const int p = (counts[e] + 63) & ~63;
      off[e] = run; pcount[e] = p; run += p;
    }
  }
}

__global__ void route_fill_kernel(const int* __restrict__ topi,
                                  const float* __restrict__ topw,
                                  const int* __restrict__ off, int* __restrict__ cursor,
                                  int* __restrict__ tokslot, float* __restrict__ wslot) {
  const int t = blockIdx.x * blockDim.x + threadIdx.x;
  if (t >= SEQ) return;
#pragma unroll
  for (int j = 0; j < 2; ++j) {
    const int e = topi[t * 2 + j];
    const int pos = atomicAdd(&cursor[e], 1);
    tokslot[off[e] + pos] = t;
    wslot[off[e] + pos] = topw[t * 2 + j];
  }
}

// ----------------------- MoE gate/up: gathered 64x64 tiles, dual-B, fused
__global__ __launch_bounds__(128)
void moe_gateup_kernel(const float* __restrict__ X, const float* __restrict__ Wg,
                       const float* __restrict__ Wu, const int* __restrict__ tokslot,
                       const float* __restrict__ wslot, const int* __restrict__ off,
                       const int* __restrict__ pcount, bf16* __restrict__ act) {
  const int e = blockIdx.x / TILES_M, tm = blockIdx.x % TILES_M;
  if (tm * 64 >= pcount[e]) return;
  const int slotBase = off[e] + tm * 64;
  __shared__ bf16 As[64][32];
  __shared__ bf16 Bg[64][32];
  __shared__ bf16 Bu[64][32];
  __shared__ int   toks[64];
  __shared__ float rws[64];
  const int tid = threadIdx.x;
  if (tid < 64) { toks[tid] = tokslot[slotBase + tid]; rws[tid] = wslot[slotBase + tid]; }
  __syncthreads();
  const int wid = tid >> 5, lane = tid & 31, lr = lane & 15, lh = lane >> 4;
  const int mw = (wid >> 1) * 32, nw = (wid & 1) * 32;
  const int n0 = blockIdx.y * 64;
  const float* WgE = Wg + (size_t)e * HDIM * IDIM;
  const float* WuE = Wu + (size_t)e * HDIM * IDIM;
  v8f ag[2][2] = {}, au[2][2] = {};
  for (int k0 = 0; k0 < HDIM; k0 += 32) {
#pragma unroll
    for (int it = 0; it < 2; ++it) {
      const int idx = tid + it * 128;
      const int r = idx >> 2, cq = (idx & 3) * 8;
      *(v8bf*)&As[r][cq] = load8_bf(&X[(size_t)toks[r] * HDIM + k0 + cq]);
      const int kr = idx >> 3, nq = (idx & 7) * 8;
      const v8bf bg = load8_bf(&WgE[(size_t)(k0 + kr) * IDIM + n0 + nq]);
      const v8bf bu = load8_bf(&WuE[(size_t)(k0 + kr) * IDIM + n0 + nq]);
#pragma unroll
      for (int j = 0; j < 8; ++j) { Bg[nq + j][kr] = bg[j]; Bu[nq + j][kr] = bu[j]; }
    }
    __syncthreads();
    const v16bf a0 = frag16s(&As[mw][0], lr, lh, 32);
    const v16bf a1 = frag16s(&As[mw + 16][0], lr, lh, 32);
    const v16bf g0 = frag16s(&Bg[nw][0], lr, lh, 32);
    const v16bf g1 = frag16s(&Bg[nw + 16][0], lr, lh, 32);
    const v16bf u0 = frag16s(&Bu[nw][0], lr, lh, 32);
    const v16bf u1 = frag16s(&Bu[nw + 16][0], lr, lh, 32);
    ag[0][0] = wmma_bf16(a0, g0, ag[0][0]);
    ag[0][1] = wmma_bf16(a0, g1, ag[0][1]);
    ag[1][0] = wmma_bf16(a1, g0, ag[1][0]);
    ag[1][1] = wmma_bf16(a1, g1, ag[1][1]);
    au[0][0] = wmma_bf16(a0, u0, au[0][0]);
    au[0][1] = wmma_bf16(a0, u1, au[0][1]);
    au[1][0] = wmma_bf16(a1, u0, au[1][0]);
    au[1][1] = wmma_bf16(a1, u1, au[1][1]);
    __syncthreads();
  }
#pragma unroll
  for (int mi = 0; mi < 2; ++mi)
#pragma unroll
    for (int ni = 0; ni < 2; ++ni)
#pragma unroll
      for (int v = 0; v < 8; ++v) {
        const int m = mw + mi * 16 + v + 8 * lh;
        const float g = ag[mi][ni][v];
        const float s = (g / (1.f + __expf(-g))) * au[mi][ni][v] * rws[m];
        act[(size_t)(slotBase + m) * IDIM + n0 + nw + ni * 16 + lr] = (bf16)s;
      }
}

// ------------------ MoE down-projection (bf16 act) + scatter-add into out
// A-tile (pure bf16 copy) staged via the Tensor Data Mover when available.
__global__ __launch_bounds__(128)
void moe_down_kernel(const bf16* __restrict__ act, const float* __restrict__ Wd,
                     const int* __restrict__ tokslot, const int* __restrict__ off,
                     const int* __restrict__ pcount, float* __restrict__ out) {
  const int e = blockIdx.x / TILES_M, tm = blockIdx.x % TILES_M;
  if (tm * 64 >= pcount[e]) return;
  const int slotBase = off[e] + tm * 64;
  __shared__ bf16 As[64][32];
  __shared__ bf16 Bs[64][32];
  __shared__ int toks[64];
  const int tid = threadIdx.x;
  if (tid < 64) toks[tid] = tokslot[slotBase + tid];
  __syncthreads();
  const int wid = tid >> 5, lane = tid & 31, lr = lane & 15, lh = lane >> 4;
  const int mw = (wid >> 1) * 32, nw = (wid & 1) * 32;
  const int n0 = blockIdx.y * 64;
  const float* WdE = Wd + (size_t)e * IDIM * HDIM;
#if HAVE_TDM
  const unsigned long long gbase =
      (unsigned long long)(size_t)act +
      (unsigned long long)slotBase * IDIM * sizeof(bf16);
  const unsigned lds_as = (unsigned)(size_t)&As[0][0];   // low 32b = LDS offset
#endif
  v8f acc[2][2] = {};
  for (int k0 = 0; k0 < IDIM; k0 += 32) {
#if HAVE_TDM
    if (wid == 0) {
      // D# per CDNA5 ISA ch.8: 2-D tensor, 64x32 bf16 tile, row stride IDIM.
      const unsigned long long ga = gbase + (unsigned long long)k0 * sizeof(bf16);
      u32x4 g0;
      g0[0] = 1u;                                            // count=1, user D#
      g0[1] = lds_as;                                        // lds_addr
      g0[2] = (unsigned)ga;                                  // global_addr[31:0]
      g0[3] = (unsigned)((ga >> 32) & 0x01FFFFFFu) | (2u << 30); // [56:32], type=2
      i32x8 g1;
      g1[0] = 0x00010000;                                    // data_size = 2B
      g1[1] = (int)((unsigned)IDIM << 16);                   // tensor_dim0 = IDIM
      g1[2] = (int)((unsigned)SLOT_CAP << 16);               // tensor_dim1
      g1[3] = (int)(32u << 16);                              // tile_dim0 = 32
      g1[4] = 64;                                            // tile_dim1 = 64
      g1[5] = IDIM;                                          // dim0_stride
      g1[6] = 0; g1[7] = 0;
      const i32x4 z4 = {0, 0, 0, 0};
      const i32x8 z8 = {0, 0, 0, 0, 0, 0, 0, 0};
      // 6-arg (clang-23 / therock) form: g0, g1, g2, g3, extra group, cpol
      __builtin_amdgcn_tensor_load_to_lds(g0, g1, z4, z4, z8, 0);
    }
#endif
#pragma unroll
    for (int it = 0; it < 2; ++it) {
      const int idx = tid + it * 128;
#if !HAVE_TDM
      const int r = idx >> 2, cq = (idx & 3) * 8;
      *(v8bf*)&As[r][cq] =
          *(const v8bf*)&act[(size_t)(slotBase + r) * IDIM + k0 + cq];
#endif
      const int kr = idx >> 3, nq = (idx & 7) * 8;
      const v8bf bb = load8_bf(&WdE[(size_t)(k0 + kr) * HDIM + n0 + nq]);
#pragma unroll
      for (int j = 0; j < 8; ++j) Bs[nq + j][kr] = bb[j];
    }
#if HAVE_TDM
    if (wid == 0) __builtin_amdgcn_s_wait_tensorcnt(0);      // TDM done on wave0
#endif
    __syncthreads();                                         // release all waves
    const v16bf a0 = frag16s(&As[mw][0], lr, lh, 32);
    const v16bf a1 = frag16s(&As[mw + 16][0], lr, lh, 32);
    const v16bf b0 = frag16s(&Bs[nw][0], lr, lh, 32);
    const v16bf b1 = frag16s(&Bs[nw + 16][0], lr, lh, 32);
    acc[0][0] = wmma_bf16(a0, b0, acc[0][0]);
    acc[0][1] = wmma_bf16(a0, b1, acc[0][1]);
    acc[1][0] = wmma_bf16(a1, b0, acc[1][0]);
    acc[1][1] = wmma_bf16(a1, b1, acc[1][1]);
    __syncthreads();
  }
#pragma unroll
  for (int mi = 0; mi < 2; ++mi)
#pragma unroll
    for (int ni = 0; ni < 2; ++ni)
#pragma unroll
      for (int v = 0; v < 8; ++v) {
        const int token = toks[mw + mi * 16 + v + 8 * lh];
        atomicAdd(&out[(size_t)token * HDIM + n0 + nw + ni * 16 + lr],
                  acc[mi][ni][v]);
      }
}

// ================================================================= host
extern "C" void kernel_launch(void* const* d_in, const int* in_sizes, int n_in,
                              void* d_out, int out_size, void* d_ws, size_t ws_size,
                              hipStream_t stream) {
  const float* hidden  = (const float*)d_in[0];
  const int*   pos     = (const int*)d_in[1];
  const float* ln1w    = (const float*)d_in[2];
  const float* wq      = (const float*)d_in[3];
  const float* bq      = (const float*)d_in[4];
  const float* wk      = (const float*)d_in[5];
  const float* bk      = (const float*)d_in[6];
  const float* wv      = (const float*)d_in[7];
  const float* bv      = (const float*)d_in[8];
  const float* wo      = (const float*)d_in[9];
  const float* bo      = (const float*)d_in[10];
  const float* ln2w    = (const float*)d_in[11];
  const float* routerw = (const float*)d_in[12];
  const float* wgate   = (const float*)d_in[13];
  const float* wup     = (const float*)d_in[14];
  const float* wdown   = (const float*)d_in[15];
  float* out = (float*)d_out;

  float* ws = (float*)d_ws;
  float* x1     = ws; ws += (size_t)SEQ * HDIM;
  float* q      = ws; ws += (size_t)SEQ * NH * HD;
  float* k      = ws; ws += (size_t)SEQ * NKV * HD;
  float* v      = ws; ws += (size_t)SEQ * NKV * HD;
  float* attno  = ws; ws += (size_t)SEQ * HDIM;
  float* hdn    = ws; ws += (size_t)SEQ * HDIM;
  float* x2     = ws; ws += (size_t)SEQ * HDIM;
  float* logits = ws; ws += (size_t)SEQ * NEXP;
  float* topw   = ws; ws += (size_t)SEQ * 2;
  float* wslot  = ws; ws += SLOT_CAP;
  bf16*  act    = (bf16*)ws; ws += (size_t)SLOT_CAP * IDIM / 2;
  int* ip      = (int*)ws;
  int* counts  = ip; ip += NEXP;
  int* cursor  = ip; ip += NEXP;
  int* off     = ip; ip += NEXP;
  int* pcount  = ip; ip += NEXP;
  int* topi    = ip; ip += SEQ * 2;
  int* tokslot = ip; ip += SLOT_CAP;

  const dim3 blk128(128);

  // ---- attention sub-block
  rmsnorm_kernel<<<SEQ, 256, 0, stream>>>(hidden, ln1w, x1, HDIM);
  gemm64<<<dim3(NH * HD / 64, SEQ / 64), blk128, 0, stream>>>(
      x1, HDIM, wq, NH * HD, bq, nullptr, q, nullptr, NH * HD, HDIM);
  gemm64<<<dim3(NKV * HD / 64, SEQ / 64), blk128, 0, stream>>>(
      x1, HDIM, wk, NKV * HD, bk, nullptr, k, nullptr, NKV * HD, HDIM);
  gemm64<<<dim3(NKV * HD / 64, SEQ / 64), blk128, 0, stream>>>(
      x1, HDIM, wv, NKV * HD, bv, nullptr, v, nullptr, NKV * HD, HDIM);
  rope_kernel<<<(SEQ * NH * (HD / 2) + 255) / 256, 256, 0, stream>>>(q, NH, pos);
  rope_kernel<<<(SEQ * NKV * (HD / 2) + 255) / 256, 256, 0, stream>>>(k, NKV, pos);
  flash_attn_kernel<<<dim3(SEQ / 64, NH), blk128, 0, stream>>>(q, k, v, attno);
  gemm64<<<dim3(HDIM / 64, SEQ / 64), blk128, 0, stream>>>(
      attno, HDIM, wo, HDIM, bo, hidden, hdn, out, HDIM, HDIM);

  // ---- MoE sub-block
  rmsnorm_kernel<<<SEQ, 256, 0, stream>>>(hdn, ln2w, x2, HDIM);
  router_kernel<<<SEQ / 16, 256, 0, stream>>>(x2, routerw, logits);
  route_init_kernel<<<(SLOT_CAP + 255) / 256, 256, 0, stream>>>(counts, cursor,
                                                                tokslot, wslot);
  route_topk_kernel<<<(SEQ + 255) / 256, 256, 0, stream>>>(logits, topi, topw,
                                                           counts);
  route_scan_kernel<<<1, 32, 0, stream>>>(counts, off, pcount);
  route_fill_kernel<<<(SEQ + 255) / 256, 256, 0, stream>>>(topi, topw, off,
                                                           cursor, tokslot, wslot);
  moe_gateup_kernel<<<dim3(NEXP * TILES_M, IDIM / 64), blk128, 0, stream>>>(
      x2, wgate, wup, tokslot, wslot, off, pcount, act);
  moe_down_kernel<<<dim3(NEXP * TILES_M, HDIM / 64), blk128, 0, stream>>>(
      act, wdown, tokslot, off, pcount, out);
}
